// Router_48352741818881
// MI455X (gfx1250) — matrix operand order repaired and checked by reading, not compiled, not run on hardware
//
#include <hip/hip_runtime.h>
#include <hip/hip_bf16.h>

typedef __attribute__((ext_vector_type(16))) _Float16 v16h;
typedef __attribute__((ext_vector_type(8)))  _Float16 v8h;
typedef __attribute__((ext_vector_type(8)))  float    v8f;

namespace {
constexpr int kB = 8192;   // batch
constexpr int kD = 4096;   // feature dim (GEMM reduction dim)
constexpr int kE = 16;     // experts
constexpr int kK = 64;     // rank per expert (GEMM N per block)

constexpr int MT   = 128;  // rows per block
constexpr int KC   = 64;   // D-chunk staged per iteration
constexpr int LSTR = 72;   // LDS row stride in halfs (64 data + 8 pad -> conflict-free b128)
}

__device__ __forceinline__ v16h cat16(v8h a, v8h b) {
  return __builtin_shufflevector(a, b, 0, 1, 2, 3, 4, 5, 6, 7,
                                       8, 9, 10, 11, 12, 13, 14, 15);
}

// D = A*B + C, f16 inputs, f32 accumulate
__device__ __forceinline__ v8f wmma_f16(v16h a, v16h b, v8f c) {
  return __builtin_amdgcn_wmma_f32_16x16x32_f16(
      /*neg_a=*/false, a, /*neg_b=*/false, b,
      /*c_mod=*/(short)0, c, /*reuse_a=*/false, /*reuse_b=*/false);
}

__global__ __launch_bounds__(256, 1) void router_rank_norm_wmma(
    const float* __restrict__ x,      // [B, D]
    const float* __restrict__ w,      // [E, K, D]
    float* __restrict__ out) {        // [B, E]
  // Staged tiles: f32 split into f16 head (h) + f16 residual (l).
  __shared__ __align__(16) _Float16 Ah[MT * LSTR];
  __shared__ __align__(16) _Float16 Al[MT * LSTR];
  __shared__ __align__(16) _Float16 Bh[kK * LSTR];
  __shared__ __align__(16) _Float16 Bl[kK * LSTR];

  const int tid     = threadIdx.x;
  const int lane    = tid & 31;
  const int wave    = tid >> 5;            // 0..7, owns 16 rows
  const int rowBase = blockIdx.x * MT;
  const int e       = blockIdx.y;
  const float* wexp = w + (size_t)e * kK * kD;

  v8f acc[4] = {};                         // 16 rows x 64 cols per wave

  const int c4 = (tid & 15) * 4;           // f32 column within chunk
  const int r0 = tid >> 4;                 // staging row start

  for (int d0 = 0; d0 < kD; d0 += KC) {
    __syncthreads();                       // protect LDS from prior reads

    // ---- stage A: x tile [128 x 64] f32 -> hi/lo f16 (fully coalesced) ----
    for (int rr = r0; rr < MT; rr += 16) {
      const float* src = x + (size_t)(rowBase + rr) * kD + (d0 + c4);
      const float4 v = *(const float4*)src;
      if (d0 + KC < kD) __builtin_prefetch(src + KC, 0, 0);  // global_prefetch_b8
      union { _Float16 h[4]; uint2 u; } ph, pl;
      const float f[4] = {v.x, v.y, v.z, v.w};
#pragma unroll
      for (int j = 0; j < 4; ++j) {
        const _Float16 h = (_Float16)f[j];
        ph.h[j] = h;
        pl.h[j] = (_Float16)(f[j] - (float)h);   // exact residual, rounded to f16
      }
      *(uint2*)&Ah[rr * LSTR + c4] = ph.u;
      *(uint2*)&Al[rr * LSTR + c4] = pl.u;
    }

    // ---- stage B: weights[e] tile [64 x 64] f32 -> hi/lo f16 ----
    for (int rr = r0; rr < kK; rr += 16) {
      const float4 v = *(const float4*)(wexp + (size_t)rr * kD + (d0 + c4));
      union { _Float16 h[4]; uint2 u; } ph, pl;
      const float f[4] = {v.x, v.y, v.z, v.w};
#pragma unroll
      for (int j = 0; j < 4; ++j) {
        const _Float16 h = (_Float16)f[j];
        ph.h[j] = h;
        pl.h[j] = (_Float16)(f[j] - (float)h);
      }
      *(uint2*)&Bh[rr * LSTR + c4] = ph.u;
      *(uint2*)&Bl[rr * LSTR + c4] = pl.u;
    }
    __syncthreads();

    // ---- compute: 2 K-steps of 32, 4 N-tiles, 3 WMMAs each (hh, hl, lh) ----
    const int mrow = wave * 16 + (lane & 15);
    const int sel  = (lane >> 4) & 1;      // lane group selects K-halves
#pragma unroll
    for (int ks = 0; ks < 2; ++ks) {
      const int kb = ks * 32;
      // A fragment: lane<16 -> K {0..7,16..23}; lane>=16 -> K {8..15,24..31}
      const _Float16* pa  = &Ah[mrow * LSTR + kb + sel * 8];
      const _Float16* pal = &Al[mrow * LSTR + kb + sel * 8];
      const v16h ah = cat16(*(const v8h*)pa,  *(const v8h*)(pa + 16));
      const v16h al = cat16(*(const v8h*)pal, *(const v8h*)(pal + 16));
#pragma unroll
      for (int t = 0; t < 4; ++t) {
        // B fragment: column n = lane&15; lane<16 -> K {0..7,16..23}; lane>=16 -> K {8..15,24..31}
        const int nrow = t * 16 + (lane & 15);
        const _Float16* pb  = &Bh[nrow * LSTR + kb + sel * 8];
        const _Float16* pbl = &Bl[nrow * LSTR + kb + sel * 8];
        const v16h bh = cat16(*(const v8h*)pb,  *(const v8h*)(pb + 16));
        const v16h bl = cat16(*(const v8h*)pbl, *(const v8h*)(pbl + 16));
        acc[t] = wmma_f16(ah, bh, acc[t]);
        acc[t] = wmma_f16(ah, bl, acc[t]);
        acc[t] = wmma_f16(al, bh, acc[t]);
      }
    }
  }

  // ---- router score: r[b,e] = sqrt(sum over 64 cols of c^2) ----
  // C layout: VGPR i, lanes 0-15 -> row i (N=lane), lanes 16-31 -> row i+8.
  const int sel  = (lane >> 4) & 1;
  const int rout = rowBase + wave * 16 + sel * 8;
#pragma unroll
  for (int i = 0; i < 8; ++i) {
    float s = acc[0][i] * acc[0][i] + acc[1][i] * acc[1][i] +
              acc[2][i] * acc[2][i] + acc[3][i] * acc[3][i];
    s += __shfl_xor(s, 1);   // reduce within 16-lane halves (wave32)
    s += __shfl_xor(s, 2);
    s += __shfl_xor(s, 4);
    s += __shfl_xor(s, 8);
    if ((lane & 15) == 0) out[(size_t)(rout + i) * kE + e] = sqrtf(s);
  }
}

extern "C" void kernel_launch(void* const* d_in, const int* in_sizes, int n_in,
                              void* d_out, int out_size, void* d_ws, size_t ws_size,
                              hipStream_t stream) {
  (void)in_sizes; (void)n_in; (void)out_size; (void)d_ws; (void)ws_size;
  const float* x = (const float*)d_in[0];   // [8192, 4096] f32
  const float* w = (const float*)d_in[1];   // [16, 64, 4096] f32
  float* out = (float*)d_out;               // [8192, 16] f32
  dim3 grid(kB / MT, kE);
  router_rank_norm_wmma<<<grid, 256, 0, stream>>>(x, w, out);
}
